// BipartPool_54717883351107
// MI455X (gfx1250) — matrix operand order repaired and checked by reading, not compiled, not run on hardware
//
#include <hip/hip_runtime.h>

// ---------------------------------------------------------------------------
// BipartPool 'attn' mode for MI455X (gfx1250, wave32, WMMA).
// Pipeline:
//   1) kv = x @ [wk;wv]^T + [bk;bv]              (bf16 out, WMMA bf16 GEMM)
//   2) qb = (xcent_base @ wq^T + bq) / sqrt(64)  (bf16 out) -- queries repeat per graph
//   3) per (graph, head): S = Q K^T, softmax, O = P V   (WMMA + shuffle softmax)
//   4) out = O_all @ wout^T + bout               (f32 out)
//   5) batchcent[i] = i / 32
// GEMM uses 128x128 block tiles, K-chunk 32, double-buffered LDS staging
// (global->reg loads for chunk c+1 overlap WMMA on chunk c).
// ---------------------------------------------------------------------------

typedef __attribute__((ext_vector_type(16))) __bf16 v16bf;
typedef __attribute__((ext_vector_type(8)))  float  v8f;

union Frag16 { v16bf v; unsigned u[8]; __bf16 h[16]; };

static constexpr int Bg    = 32;     // graphs
static constexpr int RATIO = 32;     // queries per graph
static constexpr int H     = 8;      // heads
static constexpr int E     = 512;    // embed dim
static constexpr int HD    = 64;     // head dim
static constexpr int NPG   = 512;    // nodes per graph
static constexpr int Nn    = Bg * NPG;    // 16384 nodes
static constexpr int L     = Bg * RATIO;  // 1024 queries

__device__ __forceinline__ __bf16 f2bf(float f) {
  unsigned u = __builtin_bit_cast(unsigned, f);
  u += 0x7FFFu + ((u >> 16) & 1u);               // round-to-nearest-even
  unsigned short s = (unsigned short)(u >> 16);
  return __builtin_bit_cast(__bf16, s);
}

// A-matrix (16x32 bf16) dword d -> K index (contiguous order, matches B)
__device__ __forceinline__ int amapK(int d, int half) {
  return 2 * d + 16 * half;
}
// B-matrix (32x16 bf16) dword d -> K index
__device__ __forceinline__ int bmapK(int d, int half) {
  return 2 * d + 16 * half;
}

// ---------------------------------------------------------------------------
// Generic GEMM: C[M,Nout] = A[M,K] @ W[Nout,K]^T + bias, all f32 in,
// bf16 or f32 out, (acc+bias)*scale epilogue.
// Block tile 128x128, K-chunk 32, double-buffered LDS.
// ---------------------------------------------------------------------------
template <bool OUT_BF16>
__global__ void __launch_bounds__(256)
gemm_wmma_kernel(const float* __restrict__ A, const float* __restrict__ W,
                 const float* __restrict__ bias, void* __restrict__ Cout,
                 int M, int K, int Nout, float scale) {
  constexpr int TM = 128, TN = 128, TK = 32;
  __shared__ __bf16 lA[2][TM][TK + 2];
  __shared__ __bf16 lB[2][TN][TK + 2];

  const int t = threadIdx.x;
  const int w = t >> 5, lane = t & 31;
  const int half = lane >> 4, l16 = lane & 15;
  const int mBase = blockIdx.y * TM;
  const int nBase = blockIdx.x * TN;

  v8f acc[8] = {};
  float ra[16], rb[16];  // register staging for the next K-chunk

  auto loadChunk = [&](int kk) {
#pragma unroll
    for (int i = 0; i < 16; ++i) {
      int e = t + i * 256;
      int r = e >> 5, c = e & 31;
      int gm = mBase + r;
      ra[i] = (gm < M) ? A[(size_t)gm * K + kk + c] : 0.0f;
    }
#pragma unroll
    for (int i = 0; i < 16; ++i) {
      int e = t + i * 256;
      int r = e >> 5, c = e & 31;
      int gn = nBase + r;
      rb[i] = (gn < Nout) ? W[(size_t)gn * K + kk + c] : 0.0f;
    }
  };
  auto storeChunk = [&](int buf) {
#pragma unroll
    for (int i = 0; i < 16; ++i) {
      int e = t + i * 256;
      lA[buf][e >> 5][e & 31] = f2bf(ra[i]);
    }
#pragma unroll
    for (int i = 0; i < 16; ++i) {
      int e = t + i * 256;
      lB[buf][e >> 5][e & 31] = f2bf(rb[i]);
    }
  };

  loadChunk(0);
  storeChunk(0);
  __syncthreads();

  const int nChunks = K / TK;
  for (int c = 0; c < nChunks; ++c) {
    const int buf = c & 1;
    if (c + 1 < nChunks) loadChunk((c + 1) * TK);  // overlap with WMMA below

    // A fragment: rows w*16 .. w*16+15 of the block tile
    Frag16 af;
    {
      const __bf16* ap = &lA[buf][(w << 4) + l16][0];
#pragma unroll
      for (int d = 0; d < 8; ++d)
        af.u[d] = *(const unsigned*)(ap + amapK(d, half));
    }
    // 8 N-subtiles
#pragma unroll
    for (int ns = 0; ns < 8; ++ns) {
      Frag16 bfm;
      const __bf16* bp = &lB[buf][ns * 16 + l16][0];
#pragma unroll
      for (int d = 0; d < 8; ++d)
        bfm.u[d] = *(const unsigned*)(bp + bmapK(d, half));
      acc[ns] = __builtin_amdgcn_wmma_f32_16x16x32_bf16(
          false, af.v, false, bfm.v, (short)0, acc[ns], false, false);
    }

    if (c + 1 < nChunks) storeChunk(buf ^ 1);
    __syncthreads();
  }

  // ---- epilogue ----
#pragma unroll
  for (int ns = 0; ns < 8; ++ns) {
    int n = nBase + ns * 16 + l16;
    float bv = bias[n];
#pragma unroll
    for (int r = 0; r < 8; ++r) {
      int m = mBase + (w << 4) + r + 8 * half;
      if (m < M) {
        float v = (acc[ns][r] + bv) * scale;
        if constexpr (OUT_BF16)
          ((__bf16*)Cout)[(size_t)m * Nout + n] = f2bf(v);
        else
          ((float*)Cout)[(size_t)m * Nout + n] = v;
      }
    }
  }
}

// ---------------------------------------------------------------------------
// Attention: one workgroup per (graph b, head h). Q is pre-scaled by 1/8.
// kvbf layout: [node][0..511]=K, [node][512..1023]=V (bf16).
// ---------------------------------------------------------------------------
__global__ void __launch_bounds__(256)
attn_kernel(const __bf16* __restrict__ qbf, const __bf16* __restrict__ kvbf,
            float* __restrict__ attn_out) {
  const int b = blockIdx.x, h = blockIdx.y;
  const int t = threadIdx.x;
  const int w = t >> 5, lane = t & 31;
  const int half = lane >> 4, l16 = lane & 15;

  __shared__ __bf16 sP[32][512 + 8];  // probabilities (padded: conflict-free frags)
  __shared__ float sRed[32][8];       // per-wave row partials
  __shared__ float sRow[32];          // row max, then row sum

  // ---------------- S = Q K^T ----------------
  // wave w: nTiles [4w, 4w+4), both mTiles (rows 0..31).
  v8f s[2][4] = {};
  Frag16 qf[2][2];
#pragma unroll
  for (int mt = 0; mt < 2; ++mt)
#pragma unroll
    for (int kc = 0; kc < 2; ++kc) {
      const __bf16* base = qbf + (size_t)(mt * 16 + l16) * E + h * HD + kc * 32;
#pragma unroll
      for (int d = 0; d < 8; ++d)
        qf[mt][kc].u[d] = *(const unsigned*)(base + amapK(d, half));
    }
#pragma unroll
  for (int nl = 0; nl < 4; ++nl) {
    int nTile = (w << 2) + nl;
#pragma unroll
    for (int kc = 0; kc < 2; ++kc) {
      Frag16 kf;
      const __bf16* base =
          kvbf + (size_t)(b * NPG + nTile * 16 + l16) * (2 * E) + h * HD + kc * 32;
#pragma unroll
      for (int d = 0; d < 8; ++d)
        kf.u[d] = *(const unsigned*)(base + bmapK(d, half));
#pragma unroll
      for (int mt = 0; mt < 2; ++mt)
        s[mt][nl] = __builtin_amdgcn_wmma_f32_16x16x32_bf16(
            false, qf[mt][kc].v, false, kf.v, (short)0, s[mt][nl], false, false);
    }
  }

  // ---------------- softmax (register resident) ----------------
  float pmax[2][8];
#pragma unroll
  for (int mt = 0; mt < 2; ++mt)
#pragma unroll
    for (int r = 0; r < 8; ++r)
      pmax[mt][r] = fmaxf(fmaxf(s[mt][0][r], s[mt][1][r]),
                          fmaxf(s[mt][2][r], s[mt][3][r]));
#pragma unroll
  for (int mask = 1; mask <= 8; mask <<= 1)
#pragma unroll
    for (int mt = 0; mt < 2; ++mt)
#pragma unroll
      for (int r = 0; r < 8; ++r)
        pmax[mt][r] = fmaxf(pmax[mt][r], __shfl_xor(pmax[mt][r], mask, 32));
  if (l16 == 0)
#pragma unroll
    for (int mt = 0; mt < 2; ++mt)
#pragma unroll
      for (int r = 0; r < 8; ++r)
        sRed[mt * 16 + r + 8 * half][w] = pmax[mt][r];
  __syncthreads();
  if (t < 32) {
    float m0 = sRed[t][0];
#pragma unroll
    for (int j = 1; j < 8; ++j) m0 = fmaxf(m0, sRed[t][j]);
    sRow[t] = m0;
  }
  __syncthreads();

  float psum[2][8];
#pragma unroll
  for (int mt = 0; mt < 2; ++mt)
#pragma unroll
    for (int r = 0; r < 8; ++r) {
      float rmax = sRow[mt * 16 + r + 8 * half];
      float a = 0.0f;
#pragma unroll
      for (int nl = 0; nl < 4; ++nl) {
        float p = __expf(s[mt][nl][r] - rmax);
        s[mt][nl][r] = p;
        a += p;
      }
      psum[mt][r] = a;
    }
#pragma unroll
  for (int mask = 1; mask <= 8; mask <<= 1)
#pragma unroll
    for (int mt = 0; mt < 2; ++mt)
#pragma unroll
      for (int r = 0; r < 8; ++r)
        psum[mt][r] += __shfl_xor(psum[mt][r], mask, 32);
  if (l16 == 0)
#pragma unroll
    for (int mt = 0; mt < 2; ++mt)
#pragma unroll
      for (int r = 0; r < 8; ++r)
        sRed[mt * 16 + r + 8 * half][w] = psum[mt][r];
  __syncthreads();
  if (t < 32) {
    float a = 0.0f;
#pragma unroll
    for (int j = 0; j < 8; ++j) a += sRed[t][j];
    sRow[t] = a;
  }
  __syncthreads();

#pragma unroll
  for (int mt = 0; mt < 2; ++mt)
#pragma unroll
    for (int r = 0; r < 8; ++r) {
      int m = mt * 16 + r + 8 * half;
      float inv = 1.0f / sRow[m];
#pragma unroll
      for (int nl = 0; nl < 4; ++nl) {
        int n = ((w << 2) + nl) * 16 + l16;
        sP[m][n] = f2bf(s[mt][nl][r] * inv);
      }
    }
  __syncthreads();

  // ---------------- O = P V ----------------
  const int mt = w & 1, nt = w >> 1;  // 2 m-tiles x 4 n-tiles = 8 waves
  v8f o = {};
  for (int kc = 0; kc < 16; ++kc) {
    Frag16 pf;
    const __bf16* prow = &sP[mt * 16 + l16][kc * 32];
#pragma unroll
    for (int d = 0; d < 8; ++d)
      pf.u[d] = *(const unsigned*)(prow + amapK(d, half));
    Frag16 vf;
    {
      int n = nt * 16 + l16;
      const __bf16* vbase = kvbf + (size_t)(b * NPG) * (2 * E) + E + h * HD + n;
#pragma unroll
      for (int d = 0; d < 8; ++d) {
        int key0 = kc * 32 + 2 * d + 16 * half;
        vf.h[2 * d]     = vbase[(size_t)key0 * (2 * E)];
        vf.h[2 * d + 1] = vbase[(size_t)(key0 + 1) * (2 * E)];
      }
    }
    o = __builtin_amdgcn_wmma_f32_16x16x32_bf16(
        false, pf.v, false, vf.v, (short)0, o, false, false);
  }
#pragma unroll
  for (int r = 0; r < 8; ++r) {
    int m = mt * 16 + r + 8 * half;
    int col = h * HD + nt * 16 + l16;
    attn_out[(size_t)(b * RATIO + m) * E + col] = o[r];
  }
}

__global__ void batchcent_kernel(int* __restrict__ out) {
  int i = blockIdx.x * 256 + threadIdx.x;
  if (i < L) out[i] = i >> 5;  // i / RATIO
}

// ---------------------------------------------------------------------------
extern "C" void kernel_launch(void* const* d_in, const int* in_sizes, int n_in,
                              void* d_out, int out_size, void* d_ws, size_t ws_size,
                              hipStream_t stream) {
  (void)in_sizes; (void)n_in; (void)out_size; (void)ws_size;
  const float* x          = (const float*)d_in[0];
  // d_in[1] edge_index, d_in[2] batch: unused (block-diagonal structure exploited)
  const float* xcent_base = (const float*)d_in[3];
  const float* in_proj_w  = (const float*)d_in[4];
  const float* in_proj_b  = (const float*)d_in[5];
  const float* out_proj_w = (const float*)d_in[6];
  const float* out_proj_b = (const float*)d_in[7];

  // workspace: kv (bf16, 32MB) | qb (bf16, 32KB) | attn_out (f32, 2MB)
  __bf16* kvbf = (__bf16*)d_ws;
  __bf16* qbf  = kvbf + (size_t)Nn * (2 * E);
  float*  attn = (float*)(qbf + (size_t)RATIO * E);

  // 1) K/V projection: [16384,512] x [1024,512]^T -> bf16 [16384,1024]
  gemm_wmma_kernel<true><<<dim3((2 * E) / 128, Nn / 128), 256, 0, stream>>>(
      x, in_proj_w + (size_t)E * E, in_proj_b + E, kvbf, Nn, E, 2 * E, 1.0f);
  // 2) Q projection (queries shared across graphs), fold 1/sqrt(64)=0.125
  gemm_wmma_kernel<true><<<dim3(E / 128, 1), 256, 0, stream>>>(
      xcent_base, in_proj_w, in_proj_b, qbf, RATIO, E, E, 0.125f);
  // 3) block-diagonal attention, one WG per (graph, head)
  attn_kernel<<<dim3(Bg, H), 256, 0, stream>>>(qbf, kvbf, attn);
  // 4) output projection -> d_out (f32 [1024,512])
  gemm_wmma_kernel<false><<<dim3(E / 128, L / 128), 256, 0, stream>>>(
      attn, out_proj_w, out_proj_b, (float*)d_out, L, E, E, 1.0f);
  // 5) batchcent (second tuple output), int32 appended after xcent
  batchcent_kernel<<<(L + 255) / 256, 256, 0, stream>>>((int*)d_out + (size_t)L * E);
}
